// VanillaSelfAttention_18794776887979
// MI455X (gfx1250) — matrix operand order repaired and checked by reading, not compiled, not run on hardware
//
#include <hip/hip_runtime.h>

typedef __attribute__((ext_vector_type(16))) _Float16 v16h;
typedef __attribute__((ext_vector_type(8)))  float    v8f;

#define Bb 2
#define Zg 200
#define Xg 200
#define Cc 128
#define Hh 4
#define Pp 8
#define Nn (Zg*Xg)        /* 40000 */
#define Dd 32
#define Rr (Bb*Nn)        /* 80000 rows */
#define NT_FUSED 14       /* 224 fused output cols / 16 */
#define NT_OUT 8          /* 128 / 16 */

// ---------------------------------------------------------------------------
// Prepack: fp32 weights -> f16 WMMA B-fragments.
// B-frag layout (v_wmma_f32_16x16x32_f16): lane<16 -> col N=lane, K=0..7,16..23 ;
// lane>=16 -> col N=lane-16, K=8..15,24..31. 512 halves per 32x16 fragment.
// ---------------------------------------------------------------------------
__global__ void prepack_weights_kernel(const float* __restrict__ Wv,
                                       const float* __restrict__ Woff,
                                       const float* __restrict__ Wattn,
                                       const float* __restrict__ W1,
                                       const float* __restrict__ W2,
                                       _Float16* __restrict__ bfused,
                                       _Float16* __restrict__ w1f,
                                       _Float16* __restrict__ w2f) {
  int wid  = (blockIdx.x * blockDim.x + threadIdx.x) >> 5;
  int lane = threadIdx.x & 31;
  if (wid >= 56 + 32 + 32) return;
  int nloc  = lane & 15;
  int krow0 = (lane < 16) ? 0 : 16;
  if (wid < 56) {
    int kt = wid / NT_FUSED, nt = wid % NT_FUSED;
    int col = nt * 16 + nloc;
    _Float16* dst = bfused + (size_t)wid * 512 + lane * 16;
#pragma unroll
    for (int j = 0; j < 16; ++j) {
      int k = kt * 32 + (krow0 >> 1) + (j & 7) + ((j >> 3) << 4);
      float v;
      if (col < 128)      v = Wv[k * 128 + col];
      else if (col < 192) v = Woff[k * 64 + (col - 128)];
      else                v = Wattn[k * 32 + (col - 192)];
      dst[j] = (_Float16)v;
    }
  } else {
    int f = wid - 56;
    const float* W    = (f < 32) ? W1  : W2;
    _Float16*    dst0 = (f < 32) ? w1f : w2f;
    int fl = f & 31;
    int kt = fl / NT_OUT, nt = fl % NT_OUT;
    int col = nt * 16 + nloc;
    _Float16* dst = dst0 + (size_t)fl * 512 + lane * 16;
#pragma unroll
    for (int j = 0; j < 16; ++j) {
      int k = kt * 32 + (krow0 >> 1) + (j & 7) + ((j >> 3) << 4);
      dst[j] = (_Float16)W[k * 128 + col];
    }
  }
}

// A-frag (16x32 f16) per ISA layout from an fp32 row-major matrix:
// lane<16: row=lane,     halves = K(0..7), K(16..23)
// lane>=16: row=lane-16, halves = K(8..15), K(24..31)
__device__ __forceinline__ v16h load_a_f32(const float* __restrict__ src, int ld,
                                           int row, int kbase, int c0) {
  v16h a;
  const float* p = src + (size_t)row * ld + kbase + c0;
#pragma unroll
  for (int j = 0; j < 8; ++j) {
    a[j]     = (_Float16)p[j];
    a[j + 8] = (_Float16)p[16 + j];
  }
  return a;
}

// Same from f16 row-major
__device__ __forceinline__ v16h load_a_f16(const _Float16* __restrict__ src, int ld,
                                           int row, int kbase, int c0) {
  v16h a;
  const _Float16* p = src + (size_t)row * ld + kbase + c0;
#pragma unroll
  for (int j = 0; j < 8; ++j) { a[j] = p[j]; a[j + 8] = p[16 + j]; }
  return a;
}

// ---------------------------------------------------------------------------
// Kernel 1: fused input projections. One wave per 32-row (M=32) tile:
// two A-fragment sets share every B fragment (2 WMMAs per B load).
// Nn % 32 == 0 -> a tile never crosses the batch boundary, so all store
// bases are per-wave constants and element stores use immediate offsets.
// ---------------------------------------------------------------------------
__global__ void __launch_bounds__(128)
proj_in_kernel(const float* __restrict__ q,
               const _Float16* __restrict__ bfrag,
               const float* __restrict__ bv,
               const float* __restrict__ boff,
               const float* __restrict__ battn,
               _Float16* __restrict__ value,
               float* __restrict__ offr,
               float* __restrict__ attnr) {
  int wave = (blockIdx.x * blockDim.x + threadIdx.x) >> 5;   // 32-row tile id
  int lane = threadIdx.x & 31;
  int nloc = lane & 15;
  int c0   = (lane < 16) ? 0 : 8;
  int rbase = wave * 32;
  int mofs  = (lane < 16) ? 0 : 8;   // D-layout row base per lane half
  int b_    = rbase / Nn;            // constant over the tile
  int nbase = rbase - b_ * Nn;

  v16h a0[4], a1[4];
#pragma unroll
  for (int kt = 0; kt < 4; ++kt) {
    a0[kt] = load_a_f32(q, Cc, rbase + nloc,      kt * 32, c0);
    a1[kt] = load_a_f32(q, Cc, rbase + 16 + nloc, kt * 32, c0);
  }

  // --- segment 1: value columns 0..127 (nt 0..7) ---
#pragma unroll 1
  for (int nt = 0; nt < 8; ++nt) {
    v8f acc0 = {}, acc1 = {};
#pragma unroll
    for (int kt = 0; kt < 4; ++kt) {
      v16h b = *(const v16h*)(bfrag + (size_t)(kt * NT_FUSED + nt) * 512 + lane * 16);
      acc0 = __builtin_amdgcn_wmma_f32_16x16x32_f16(false, a0[kt], false, b,
                                                    (short)0, acc0, false, false);
      acc1 = __builtin_amdgcn_wmma_f32_16x16x32_f16(false, a1[kt], false, b,
                                                    (short)0, acc1, false, false);
    }
    int col = nt * 16 + nloc;
    int h = col >> 5, d = col & 31;
    float bias = bv[col];
    // base = value[b_, h, nbase + mofs, d]; rows advance by Dd halves
    _Float16* p0 = value + (((size_t)(b_ * Hh + h)) * Nn + nbase + mofs) * Dd + d;
#pragma unroll
    for (int i = 0; i < 8; ++i) {
      p0[i * Dd]        = (_Float16)(acc0[i] + bias);
      p0[(i + 16) * Dd] = (_Float16)(acc1[i] + bias);
    }
  }

  // --- segment 2: sampling offsets, columns 128..191 (nt 8..11) ---
#pragma unroll 1
  for (int nt = 8; nt < 12; ++nt) {
    v8f acc0 = {}, acc1 = {};
#pragma unroll
    for (int kt = 0; kt < 4; ++kt) {
      v16h b = *(const v16h*)(bfrag + (size_t)(kt * NT_FUSED + nt) * 512 + lane * 16);
      acc0 = __builtin_amdgcn_wmma_f32_16x16x32_f16(false, a0[kt], false, b,
                                                    (short)0, acc0, false, false);
      acc1 = __builtin_amdgcn_wmma_f32_16x16x32_f16(false, a1[kt], false, b,
                                                    (short)0, acc1, false, false);
    }
    int colo = nt * 16 + nloc - 128;
    float bias = boff[colo];
    float* p0 = offr + (size_t)(rbase + mofs) * 64 + colo;
#pragma unroll
    for (int i = 0; i < 8; ++i) {
      p0[i * 64]        = acc0[i] + bias;
      p0[(i + 16) * 64] = acc1[i] + bias;
    }
  }

  // --- segment 3: attention logits, columns 192..223 (nt 12..13) ---
#pragma unroll 1
  for (int nt = 12; nt < 14; ++nt) {
    v8f acc0 = {}, acc1 = {};
#pragma unroll
    for (int kt = 0; kt < 4; ++kt) {
      v16h b = *(const v16h*)(bfrag + (size_t)(kt * NT_FUSED + nt) * 512 + lane * 16);
      acc0 = __builtin_amdgcn_wmma_f32_16x16x32_f16(false, a0[kt], false, b,
                                                    (short)0, acc0, false, false);
      acc1 = __builtin_amdgcn_wmma_f32_16x16x32_f16(false, a1[kt], false, b,
                                                    (short)0, acc1, false, false);
    }
    int cola = nt * 16 + nloc - 192;
    float bias = battn[cola];
    float* p0 = attnr + (size_t)(rbase + mofs) * 32 + cola;
#pragma unroll
    for (int i = 0; i < 8; ++i) {
      p0[i * 32]        = acc0[i] + bias;
      p0[(i + 16) * 32] = acc1[i] + bias;
    }
  }
}

// ---------------------------------------------------------------------------
// Kernel 2: deformable bilinear sampling + softmax point reduce (branchless).
// One wave per (b,n,h); lane = channel d. Pixel coords reduce to
// ix = z + off_x, iy = x + off_y. Out-of-bounds corners contribute via
// zeroed weights; indices always clamped in-bounds (reference semantics).
// ---------------------------------------------------------------------------
__global__ void deform_sample_kernel(const float* __restrict__ offr,
                                     const float* __restrict__ attnr,
                                     const _Float16* __restrict__ value,
                                     _Float16* __restrict__ hidden) {
  int w    = (blockIdx.x * blockDim.x + threadIdx.x) >> 5;   // (r, h) index
  int lane = threadIdx.x & 31;
  int h  = w & (Hh - 1);
  int r  = w >> 2;
  int b_ = r / Nn, n = r - b_ * Nn;
  int z  = n / Xg, x = n - z * Xg;

  // Every subgroup of 8 lanes replicates the 8 points (p = lane&7).
  int p = lane & 7;
  float ox = offr[(size_t)r * 64 + (h * Pp + p) * 2 + 0];
  float oy = offr[(size_t)r * 64 + (h * Pp + p) * 2 + 1];
  float lg = attnr[(size_t)r * 32 + h * Pp + p];

  float mx = lg;
#pragma unroll
  for (int o = 4; o > 0; o >>= 1) mx = fmaxf(mx, __shfl_xor(mx, o, 8));
  float e = __expf(lg - mx);
  float s = e;
#pragma unroll
  for (int o = 4; o > 0; o >>= 1) s += __shfl_xor(s, o, 8);
  float wgt = e / s;
  float ixl = (float)z + ox;
  float iyl = (float)x + oy;

  const _Float16* img = value + ((size_t)(b_ * Hh + h) * Nn) * Dd + lane;
  float acc = 0.f;
#pragma unroll 1
  for (int pp = 0; pp < Pp; ++pp) {
    float wp = __shfl(wgt, pp, 8);
    float ix = __shfl(ixl, pp, 8);
    float iy = __shfl(iyl, pp, 8);
    float x0f = floorf(ix), y0f = floorf(iy);
    float lx = ix - x0f, ly = iy - y0f;
    int x0 = (int)x0f, y0 = (int)y0f;
    int x1 = x0 + 1, y1 = y0 + 1;
    // validity folded into weights -> branchless gathers on clamped indices
    float fx0 = ((x0 >= 0) & (x0 < Xg)) ? 1.f : 0.f;
    float fx1 = ((x1 >= 0) & (x1 < Xg)) ? 1.f : 0.f;
    float fy0 = ((y0 >= 0) & (y0 < Zg)) ? 1.f : 0.f;
    float fy1 = ((y1 >= 0) & (y1 < Zg)) ? 1.f : 0.f;
    float w00 = (1.f - lx) * (1.f - ly) * fx0 * fy0;
    float w10 = lx * (1.f - ly) * fx1 * fy0;
    float w01 = (1.f - lx) * ly * fx0 * fy1;
    float w11 = lx * ly * fx1 * fy1;
    int x0c = min(max(x0, 0), Xg - 1), x1c = min(max(x1, 0), Xg - 1);
    int y0c = min(max(y0, 0), Zg - 1), y1c = min(max(y1, 0), Zg - 1);
    const _Float16* row0 = img + (size_t)(y0c * Xg) * Dd;
    const _Float16* row1 = img + (size_t)(y1c * Xg) * Dd;
    float v00 = (float)row0[(size_t)x0c * Dd];
    float v10 = (float)row0[(size_t)x1c * Dd];
    float v01 = (float)row1[(size_t)x0c * Dd];
    float v11 = (float)row1[(size_t)x1c * Dd];
    acc += wp * (w00 * v00 + w10 * v10 + w01 * v01 + w11 * v11);
  }
  hidden[(size_t)r * Cc + h * Dd + lane] = (_Float16)acc;
}

// ---------------------------------------------------------------------------
// Kernel 3: fused double output projection with residual, M=32 per wave.
// LDS staging transposes GEMM1 D-layout -> GEMM2 A-layout.
// ---------------------------------------------------------------------------
__global__ void __launch_bounds__(128)
proj_out_kernel(const _Float16* __restrict__ hidden,
                const _Float16* __restrict__ w1f,
                const _Float16* __restrict__ w2f,
                const float* __restrict__ b1,
                const float* __restrict__ b2,
                const float* __restrict__ q,
                float* __restrict__ out) {
  __shared__ _Float16 stage[4][32 * 136];   // 4 waves * 32x128 tile, padded
  int wslot = threadIdx.x >> 5;
  int wave  = (blockIdx.x * blockDim.x + threadIdx.x) >> 5;
  int lane  = threadIdx.x & 31;
  int nloc  = lane & 15;
  int c0    = (lane < 16) ? 0 : 8;
  int mofs  = (lane < 16) ? 0 : 8;
  int rbase = wave * 32;

  v16h a0[4], a1[4];
#pragma unroll
  for (int kt = 0; kt < 4; ++kt) {
    a0[kt] = load_a_f16(hidden, Cc, rbase + nloc,      kt * 32, c0);
    a1[kt] = load_a_f16(hidden, Cc, rbase + 16 + nloc, kt * 32, c0);
  }

  // GEMM1 + bias1 -> LDS tile (f16)
#pragma unroll 1
  for (int nt = 0; nt < NT_OUT; ++nt) {
    v8f acc0 = {}, acc1 = {};
#pragma unroll
    for (int kt = 0; kt < 4; ++kt) {
      v16h b = *(const v16h*)(w1f + (size_t)(kt * NT_OUT + nt) * 512 + lane * 16);
      acc0 = __builtin_amdgcn_wmma_f32_16x16x32_f16(false, a0[kt], false, b,
                                                    (short)0, acc0, false, false);
      acc1 = __builtin_amdgcn_wmma_f32_16x16x32_f16(false, a1[kt], false, b,
                                                    (short)0, acc1, false, false);
    }
    int col = nt * 16 + nloc;
    float bias = b1[col];
    _Float16* sp = &stage[wslot][mofs * 136 + col];
#pragma unroll
    for (int i = 0; i < 8; ++i) {
      sp[i * 136]        = (_Float16)(acc0[i] + bias);
      sp[(i + 16) * 136] = (_Float16)(acc1[i] + bias);
    }
  }
  __syncthreads();

  v16h a2_0[4], a2_1[4];
#pragma unroll
  for (int kt = 0; kt < 4; ++kt) {
    {
      v16h t;
      const _Float16* ptr = &stage[wslot][nloc * 136 + kt * 32 + c0];
#pragma unroll
      for (int j = 0; j < 8; ++j) { t[j] = ptr[j]; t[j + 8] = ptr[16 + j]; }
      a2_0[kt] = t;
    }
    {
      v16h t;
      const _Float16* ptr = &stage[wslot][(nloc + 16) * 136 + kt * 32 + c0];
#pragma unroll
      for (int j = 0; j < 8; ++j) { t[j] = ptr[j]; t[j + 8] = ptr[16 + j]; }
      a2_1[kt] = t;
    }
  }

  // GEMM2 + bias2 + residual -> fp32 output
#pragma unroll 1
  for (int nt = 0; nt < NT_OUT; ++nt) {
    v8f acc0 = {}, acc1 = {};
#pragma unroll
    for (int kt = 0; kt < 4; ++kt) {
      v16h b = *(const v16h*)(w2f + (size_t)(kt * NT_OUT + nt) * 512 + lane * 16);
      acc0 = __builtin_amdgcn_wmma_f32_16x16x32_f16(false, a2_0[kt], false, b,
                                                    (short)0, acc0, false, false);
      acc1 = __builtin_amdgcn_wmma_f32_16x16x32_f16(false, a2_1[kt], false, b,
                                                    (short)0, acc1, false, false);
    }
    int col = nt * 16 + nloc;
    float bias = b2[col];
    const float* qp = q + (size_t)(rbase + mofs) * Cc + col;
    float*       op = out + (size_t)(rbase + mofs) * Cc + col;
#pragma unroll
    for (int i = 0; i < 8; ++i) {
      op[i * Cc]        = acc0[i] + bias + qp[i * Cc];
      op[(i + 16) * Cc] = acc1[i] + bias + qp[(i + 16) * Cc];
    }
  }
}

// ---------------------------------------------------------------------------
extern "C" void kernel_launch(void* const* d_in, const int* in_sizes, int n_in,
                              void* d_out, int out_size, void* d_ws, size_t ws_size,
                              hipStream_t stream) {
  const float* q     = (const float*)d_in[0];
  const float* Wv    = (const float*)d_in[1];
  const float* bv    = (const float*)d_in[2];
  const float* Woff  = (const float*)d_in[3];
  const float* boff  = (const float*)d_in[4];
  const float* Wattn = (const float*)d_in[5];
  const float* battn = (const float*)d_in[6];
  const float* W1    = (const float*)d_in[7];
  const float* b1    = (const float*)d_in[8];
  const float* W2    = (const float*)d_in[9];
  const float* b2    = (const float*)d_in[10];
  float* out = (float*)d_out;

  char* ws = (char*)d_ws;
  // Workspace layout (all offsets 256B-aligned):
  size_t off_bfused = 0;                                           // 57344 B
  size_t off_w1f    = off_bfused + 56 * 512 * 2;                   // 57344
  size_t off_w2f    = off_w1f + 32 * 512 * 2;                      // 90112
  size_t off_value  = off_w2f + 32 * 512 * 2;                      // 122880
  size_t off_offr   = off_value + (size_t)Bb * Hh * Nn * Dd * 2;   // 20,602,880
  size_t off_attnr  = off_offr + (size_t)Rr * 64 * 4;              // 41,082,880
  size_t off_hid    = off_attnr + (size_t)Rr * 32 * 4;             // 51,322,880
  _Float16* bfused = (_Float16*)(ws + off_bfused);
  _Float16* w1f    = (_Float16*)(ws + off_w1f);
  _Float16* w2f    = (_Float16*)(ws + off_w2f);
  _Float16* value  = (_Float16*)(ws + off_value);
  float*    offr   = (float*)(ws + off_offr);
  float*    attnr  = (float*)(ws + off_attnr);
  _Float16* hidden = (_Float16*)(ws + off_hid);

  // 1) prepack weights into WMMA B-fragments (120 waves)
  prepack_weights_kernel<<<15, 256, 0, stream>>>(Wv, Woff, Wattn, W1, W2,
                                                 bfused, w1f, w2f);
  // 2) fused input projections: 2500 M=32 tiles, 4 waves/block
  proj_in_kernel<<<Rr / 32 / 4, 128, 0, stream>>>(q, bfused, bv, boff, battn,
                                                  value, offr, attnr);
  // 3) deformable sampling: one wave per (b,n,h) = 320000 waves
  deform_sample_kernel<<<(Rr * Hh) / 8, 256, 0, stream>>>(offr, attnr, value,
                                                          hidden);
  // 4) fused double output projection + residual
  proj_out_kernel<<<Rr / 32 / 4, 128, 0, stream>>>(hidden, w1f, w2f, b1, b2, q,
                                                   out);
}
